// MaterializedAttentionDistancePredictor_23811298690009
// MI455X (gfx1250) — matrix operand order repaired and checked by reading, not compiled, not run on hardware
//
#include <hip/hip_runtime.h>
#include <hip/hip_bf16.h>

// CDNA5 / gfx1250 wave32 WMMA types
typedef __attribute__((ext_vector_type(16))) _Float16 v16h;
typedef __attribute__((ext_vector_type(8)))  _Float16 v8h;
typedef __attribute__((ext_vector_type(8)))  float    v8f;

// Problem constants
#define B_   2
#define L_   256
#define CIN_ 20
#define C_   64
#define H_   8
#define AS_  64
#define VS_  64
#define D_   12
#define N1   512       // B*L
#define NLL  65536     // L*L
#define N2   131072    // B*L*L

__device__ __forceinline__ v8h ldg8(const _Float16* p) { return *(const v8h*)p; }
__device__ __forceinline__ v16h cat16(v8h lo, v8h hi) {
  return __builtin_shufflevector(lo, hi, 0, 1, 2, 3, 4, 5, 6, 7, 8, 9, 10, 11, 12, 13, 14, 15);
}

// ======================= staging / elementwise kernels =======================

// src f32 [K][N] -> dst f16 [N][K]  (write-coalesced transpose + convert)
__global__ void cvt_transpose_kernel(const float* __restrict__ s, _Float16* __restrict__ d,
                                     int K, int N) {
  int i = blockIdx.x * blockDim.x + threadIdx.x;
  if (i >= K * N) return;
  int n = i / K, k = i % K;
  d[i] = (_Float16)s[(size_t)k * N + n];
}

// Zero-pad K dim of weight W[O][K] (leading dim ld, col offset col0) -> Wh[O][Kp] f16
__global__ void pad_weight_kernel(const float* __restrict__ W, int ld, int col0,
                                  _Float16* __restrict__ Wh, int O, int K, int Kp) {
  int i = blockIdx.x * blockDim.x + threadIdx.x;
  if (i >= O * Kp) return;
  int o = i / Kp, k = i % Kp;
  Wh[i] = (k < K) ? (_Float16)W[(size_t)o * ld + col0 + k] : (_Float16)0.0f;
}

// inputs[b][c][l] -> Xh[b*L+l][32] (K-padded, transposed for B operand)
__global__ void build_h_inputs_kernel(const float* __restrict__ in, _Float16* __restrict__ xh) {
  int i = blockIdx.x * blockDim.x + threadIdx.x;
  if (i >= N1 * 32) return;
  int n = i / 32, c = i % 32, b = n >> 8, l = n & 255;
  float v = (c < CIN_) ? in[((size_t)b * CIN_ + c) * L_ + l] : 0.0f;
  xh[i] = (_Float16)v;
}

// data = concat(nodes, inputs) -> hd[b*L+l][96] f16 (cols 84..95 zero)
__global__ void build_h_data_kernel(const float* __restrict__ nodes, const float* __restrict__ in,
                                    _Float16* __restrict__ hd) {
  int i = blockIdx.x * blockDim.x + threadIdx.x;
  if (i >= N1 * 96) return;
  int n = i / 96, c = i % 96, b = n >> 8, l = n & 255;
  float v = 0.0f;
  if (c < C_) v = nodes[(size_t)c * N1 + n];
  else if (c < C_ + CIN_) v = in[((size_t)b * CIN_ + (c - C_)) * L_ + l];
  hd[i] = (_Float16)v;
}

// ppd[o][d] = sum_c Wp[o][c] * pe(c, d)   (positional term depends only on |l-m|)
__global__ void ppd_kernel(const float* __restrict__ edge_w, float* __restrict__ ppd) {
  int i = blockIdx.x * blockDim.x + threadIdx.x;
  if (i >= C_ * L_) return;
  int o = i >> 8, d = i & 255;
  const float* wp = edge_w + (size_t)o * 268 + 168;
  float fd = (float)d, s = 0.0f;
  for (int c = 0; c < 50; ++c) {
    float f = fd / __powf(10000.0f, 0.02f * (float)c);
    s += wp[c] * __sinf(f) + wp[50 + c] * __cosf(f);
  }
  ppd[i] = s;
}

// edges[c][b][l][m] = ex[c][b*L+m] + ey[c][b*L+l] + ppd[c][|l-m|]
__global__ void build_edges_kernel(const float* __restrict__ ex, const float* __restrict__ ey,
                                   const float* __restrict__ ppd, float* __restrict__ edges) {
  int i = blockIdx.x * blockDim.x + threadIdx.x;
  if (i >= C_ * N2) return;
  int c = i >> 17;
  int r = i & (N2 - 1);
  int b = r >> 16, lm = r & 65535, l = lm >> 8, m = lm & 255;
  int dd = (l > m) ? (l - m) : (m - l);
  edges[i] = ex[(size_t)c * N1 + b * L_ + m] + ey[(size_t)c * N1 + b * L_ + l] + ppd[c * L_ + dd];
}

// t[n][c] = edges[c][n] + ni[c][b*L+l] + nj[c][b*L+m]   (transposed f16 out, coalesced writes)
__global__ void build_t_kernel(const float* __restrict__ edges, const float* __restrict__ ni,
                               const float* __restrict__ nj, _Float16* __restrict__ th) {
  int i = blockIdx.x * blockDim.x + threadIdx.x;
  if (i >= C_ * N2) return;
  int n = i >> 6, c = i & 63;
  int b = n >> 16, lm = n & 65535, l = lm >> 8, m = lm & 255;
  th[i] = (_Float16)(edges[(size_t)c * N2 + n] + ni[(size_t)c * N1 + b * L_ + l] +
                     nj[(size_t)c * N1 + b * L_ + m]);
}

// px[c][b*L+l] = mean_m edges[c][b][l][m]
__global__ void mean_last_kernel(const float* __restrict__ edges, float* __restrict__ px) {
  int i = blockIdx.x * blockDim.x + threadIdx.x;
  if (i >= C_ * N1) return;
  int c = i / N1, n = i % N1, b = n >> 8, l = n & 255;
  const float* p = edges + ((size_t)c * B_ + b) * NLL + (size_t)l * L_;
  float s = 0.0f;
  for (int m = 0; m < L_; ++m) s += p[m];
  px[i] = s * (1.0f / L_);
}

// py[c][b*L+m] = mean_l edges[c][b][l][m]
__global__ void mean_mid_kernel(const float* __restrict__ edges, float* __restrict__ py) {
  int i = blockIdx.x * blockDim.x + threadIdx.x;
  if (i >= C_ * N1) return;
  int c = i / N1, n = i % N1, b = n >> 8, m = n & 255;
  const float* p = edges + ((size_t)c * B_ + b) * NLL + m;
  float s = 0.0f;
  for (int l = 0; l < L_; ++l) s += p[(size_t)l * L_];
  py[i] = s * (1.0f / L_);
}

// dilated conv1d k=15 pad=7*dil; optional relu; optional residual Y = resid + alpha*conv
__global__ void conv1d_k15_kernel(const float* __restrict__ X, const float* __restrict__ W,
                                  const float* __restrict__ bias, float* Y,
                                  int IC, int OC, int dil, int relu,
                                  const float* resid, const float* __restrict__ alpha_p) {
  int idx = blockIdx.x * blockDim.x + threadIdx.x;
  if (idx >= OC * N1) return;
  int oc = idx / N1, n = idx % N1, b = n >> 8, l = n & 255;
  float s = bias[oc];
  for (int ic = 0; ic < IC; ++ic) {
    const float* wr = W + ((size_t)oc * IC + ic) * 15;
    const float* xr = X + (size_t)ic * N1 + b * L_;
#pragma unroll
    for (int t = 0; t < 15; ++t) {
      int ll = l + (t - 7) * dil;
      if (ll >= 0 && ll < L_) s += wr[t] * xr[ll];
    }
  }
  if (relu) s = fmaxf(s, 0.0f);
  if (resid) s = resid[idx] + (*alpha_p) * s;
  Y[idx] = s;
}

// softmax over last dim (256); one wave per row; writes f16 attn
__global__ void softmax_kernel(const float* __restrict__ logits, _Float16* __restrict__ attn_h) {
  int row = blockIdx.x;
  int lane = threadIdx.x;
  const float* src = logits + (size_t)row * L_;
  float vals[8];
  float mx = -3.0e38f;
#pragma unroll
  for (int i = 0; i < 8; ++i) { vals[i] = src[lane + 32 * i]; mx = fmaxf(mx, vals[i]); }
  for (int off = 16; off > 0; off >>= 1) mx = fmaxf(mx, __shfl_xor(mx, off));
  float sum = 0.0f;
#pragma unroll
  for (int i = 0; i < 8; ++i) { vals[i] = __expf(vals[i] - mx); sum += vals[i]; }
  for (int off = 16; off > 0; off >>= 1) sum += __shfl_xor(sum, off);
  float inv = 1.0f / sum;
  _Float16* dst = attn_h + (size_t)row * L_;
#pragma unroll
  for (int i = 0; i < 8; ++i) dst[lane + 32 * i] = (_Float16)(vals[i] * inv);
}

// ======================= WMMA GEMM kernels =======================
// Y[M x N] = A[M][K](f16) * Bt[N][K](f16, transposed); wave computes 16M x 64N (4 tiles).
// store_mode: 0 row-major [row][col] (+resid / f16 out, optionally transposed f16 out)
//             1 seq head out[b][row][l], col = b*256+l
//             2 pair head out[b][row][l][m], col = b*65536+l*256+m
__global__ __launch_bounds__(32)
void wmma_gemm_kernel(const _Float16* __restrict__ A, const _Float16* __restrict__ Bt,
                      int M, int N, int K,
                      const float* __restrict__ bias, int relu,
                      float* outF, _Float16* outH, int outHT,
                      const float* resid, const float* __restrict__ ralpha,
                      int store_mode) {
  const int mt = blockIdx.x * 16;
  const int nt = blockIdx.y * 64;
  const int lane = threadIdx.x;
  const int g = lane >> 4, t = lane & 15;
  int arow = mt + t; if (arow > M - 1) arow = M - 1;   // clamp loads; stores guarded
  const _Float16* ap = A + (size_t)arow * K;
  const _Float16* bp = Bt + (size_t)(nt + t) * K;      // col tile j adds 16*j*K
  v8f acc[4] = {{}, {}, {}, {}};
  for (int k0 = 0; k0 < K; k0 += 32) {
    v16h af = cat16(ldg8(ap + k0 + g * 16), ldg8(ap + k0 + g * 16 + 8));
    const int kb = k0 + g * 16;
#pragma unroll
    for (int j = 0; j < 4; ++j) {
      v16h bf = cat16(ldg8(bp + (size_t)16 * j * K + kb), ldg8(bp + (size_t)16 * j * K + kb + 8));
      acc[j] = __builtin_amdgcn_wmma_f32_16x16x32_f16(false, af, false, bf, (short)0, acc[j],
                                                      false, false);
    }
  }
  float aval = 1.0f;
  if (ralpha) aval = *ralpha;
#pragma unroll
  for (int j = 0; j < 4; ++j) {
    const int col = nt + 16 * j + t;
    if (outHT && (mt + 16 <= M)) {  // packed transposed f16 store: 8 consecutive rows per lane
      v8h pk;
#pragma unroll
      for (int r = 0; r < 8; ++r) {
        float v = acc[j][r];
        if (bias) v += bias[mt + r + 8 * g];
        if (relu) v = fmaxf(v, 0.0f);
        pk[r] = (_Float16)v;
      }
      *(v8h*)(outH + (size_t)col * M + mt + 8 * g) = pk;
      continue;
    }
#pragma unroll
    for (int r = 0; r < 8; ++r) {
      int row = mt + r + 8 * g;
      if (row < M) {
        float v = acc[j][r];
        if (bias) v += bias[row];
        if (relu) v = fmaxf(v, 0.0f);
        if (store_mode == 0) {
          size_t idx = (size_t)row * N + col;
          if (resid) v = resid[idx] + aval * v;
          if (outF) outF[idx] = v;
          if (outH && !outHT) outH[idx] = (_Float16)v;
        } else if (store_mode == 1) {
          int b = col >> 8, l = col & 255;
          outF[((size_t)b * M + row) * L_ + l] = v;
        } else {
          int b = col >> 16, lm = col & 65535;
          outF[((size_t)b * M + row) * NLL + lm] = v;
        }
      }
    }
  }
}

// logits[b][h][l][m] = (q^T k)/sqrt(AS) + biasY[h][b,l,m], masked on m.
// qT,kT: [b*L + n][512] transposed f16 (contiguous K per lane). Wave: 16 l x 64 m.
__global__ __launch_bounds__(32)
void attn_logits_kernel(const _Float16* __restrict__ qT, const _Float16* __restrict__ kT,
                        const float* __restrict__ biasY, const unsigned char* __restrict__ mask,
                        float* __restrict__ logits) {
  const int bh = blockIdx.x;
  const int b = bh >> 3, h = bh & 7;
  const int lt = blockIdx.y * 16, mt0 = blockIdx.z * 64;
  const int lane = threadIdx.x;
  const int g = lane >> 4, t = lane & 15;
  const _Float16* qp = qT + (size_t)(b * L_ + lt + t) * 512 + h * AS_;
  const _Float16* kp = kT + (size_t)(b * L_ + mt0 + t) * 512 + h * AS_;
  v8f acc[4] = {{}, {}, {}, {}};
#pragma unroll
  for (int k0 = 0; k0 < AS_; k0 += 32) {
    v16h af = cat16(ldg8(qp + k0 + g * 16), ldg8(qp + k0 + g * 16 + 8));
    const int kb = k0 + g * 16;
#pragma unroll
    for (int j = 0; j < 4; ++j) {
      v16h bf = cat16(ldg8(kp + (size_t)16 * j * 512 + kb), ldg8(kp + (size_t)16 * j * 512 + kb + 8));
      acc[j] = __builtin_amdgcn_wmma_f32_16x16x32_f16(false, af, false, bf, (short)0, acc[j],
                                                      false, false);
    }
  }
#pragma unroll
  for (int j = 0; j < 4; ++j) {
    const int m = mt0 + 16 * j + t;
    const int valid = mask[b * L_ + m];
#pragma unroll
    for (int r = 0; r < 8; ++r) {
      int l = lt + r + 8 * g;
      size_t colidx = (size_t)b * NLL + (size_t)l * L_ + m;
      float v = acc[j][r] * 0.125f + biasY[(size_t)h * N2 + colidx];
      if (!valid) v = -1.0e9f;
      logits[(((size_t)b * H_ + h) * NLL) + (size_t)l * L_ + m] = v;
    }
  }
}

// ctxT[b*L+l][h*VS+v] = sum_m v[h*VS+v][b*L+m] * attn[b][h][l][m]  (transposed f16, packed stores)
__global__ __launch_bounds__(32)
void attn_ctx_kernel(const _Float16* __restrict__ ah, const _Float16* __restrict__ vh,
                     _Float16* __restrict__ ctxT) {
  const int bh = blockIdx.x;
  const int b = bh >> 3, h = bh & 7;
  const int vt = blockIdx.y * 16, lt = blockIdx.z * 64;
  const int lane = threadIdx.x;
  const int g = lane >> 4, t = lane & 15;
  const _Float16* vrow = vh + (size_t)(h * VS_ + vt + t) * N1 + b * L_;
  const _Float16* arow = ah + (((size_t)b * H_ + h) * NLL) + (size_t)(lt + t) * L_;
  v8f acc[4] = {{}, {}, {}, {}};
  for (int k0 = 0; k0 < L_; k0 += 32) {
    v16h af = cat16(ldg8(vrow + k0 + g * 16), ldg8(vrow + k0 + g * 16 + 8));
    const int kb = k0 + g * 16;
#pragma unroll
    for (int j = 0; j < 4; ++j) {
      v16h bf = cat16(ldg8(arow + (size_t)16 * j * L_ + kb), ldg8(arow + (size_t)16 * j * L_ + kb + 8));
      acc[j] = __builtin_amdgcn_wmma_f32_16x16x32_f16(false, af, false, bf, (short)0, acc[j],
                                                      false, false);
    }
  }
#pragma unroll
  for (int j = 0; j < 4; ++j) {
    const int l = lt + 16 * j + t;
    v8h pk;
#pragma unroll
    for (int r = 0; r < 8; ++r) pk[r] = (_Float16)acc[j][r];
    *(v8h*)(ctxT + (size_t)(b * L_ + l) * 512 + h * VS_ + vt + 8 * g) = pk;
  }
}

// ======================= host orchestration =======================

extern "C" void kernel_launch(void* const* d_in, const int* in_sizes, int n_in,
                              void* d_out, int out_size, void* d_ws, size_t ws_size,
                              hipStream_t stream) {
  (void)in_sizes; (void)n_in; (void)out_size; (void)ws_size;

  // Input indexing: depth-first insertion order of setup_inputs()
  const float*         inputs = (const float*)d_in[0];
  const unsigned char* mask   = (const unsigned char*)d_in[1];
  const float*         pre_w  = (const float*)d_in[2];
  const float*         pre_b  = (const float*)d_in[3];
  const float* edge_w = (const float*)d_in[74];
  const float* wq  = (const float*)d_in[75];
  const float* wk  = (const float*)d_in[76];
  const float* wv  = (const float*)d_in[77];
  const float* wb  = (const float*)d_in[78];
  const float* wo  = (const float*)d_in[79];
  const float* bo  = (const float*)d_in[80];
  const float* wi  = (const float*)d_in[81];
  const float* wj  = (const float*)d_in[82];
  const float* we1 = (const float*)d_in[83];
  const float* we2 = (const float*)d_in[84];
  const float* a_n = (const float*)d_in[85];
  const float* a_e = (const float*)d_in[86];

  // Workspace carve-out (256B aligned slices)
  char* ws = (char*)d_ws;
  size_t off = 0;
  auto alloc = [&](size_t bytes) -> void* {
    void* p = ws + off;
    off += (bytes + 255) & ~(size_t)255;
    return p;
  };
  float*    nodes   = (float*)alloc((size_t)C_ * N1 * 4);
  float*    h1      = (float*)alloc((size_t)32 * N1 * 4);
  float*    h2      = (float*)alloc((size_t)32 * N1 * 4);
  _Float16* h1h     = (_Float16*)alloc((size_t)N1 * 32 * 2);    // [N1][32]
  _Float16* h_in    = (_Float16*)alloc((size_t)N1 * 32 * 2);    // [N1][32]
  _Float16* h_data  = (_Float16*)alloc((size_t)N1 * 96 * 2);    // [N1][96]
  _Float16* wtmp    = (_Float16*)alloc((size_t)512 * 64 * 2);
  float*    exf     = (float*)alloc((size_t)C_ * N1 * 4);
  float*    eyf     = (float*)alloc((size_t)C_ * N1 * 4);
  float*    ppd     = (float*)alloc((size_t)C_ * L_ * 4);
  float*    edges   = (float*)alloc((size_t)C_ * N2 * 4);       // [c][b][l][m]
  _Float16* h_edges = (_Float16*)alloc((size_t)N2 * C_ * 2);    // [n][64]
  _Float16* qT      = (_Float16*)alloc((size_t)N1 * 512 * 2);   // [n][512]
  _Float16* kT      = (_Float16*)alloc((size_t)N1 * 512 * 2);   // [n][512]
  _Float16* vh      = (_Float16*)alloc((size_t)512 * N1 * 2);   // [row][n]
  float*    biasY   = (float*)alloc((size_t)H_ * N2 * 4);
  float*    logits  = (float*)alloc((size_t)B_ * H_ * NLL * 4);
  _Float16* attn_h  = (_Float16*)alloc((size_t)B_ * H_ * NLL * 2);
  _Float16* ctxT    = (_Float16*)alloc((size_t)N1 * 512 * 2);   // [n][512]
  _Float16* nodes_h = (_Float16*)alloc((size_t)N1 * C_ * 2);    // [n][64]
  float*    nif     = (float*)alloc((size_t)C_ * N1 * 4);
  float*    njf     = (float*)alloc((size_t)C_ * N1 * 4);
  _Float16* t_h     = (_Float16*)alloc((size_t)N2 * C_ * 2);    // [n][64]
  _Float16* hbufT   = (_Float16*)alloc((size_t)N2 * C_ * 2);    // [n][64]
  float*    pxf     = (float*)alloc((size_t)C_ * N1 * 4);
  float*    pyf     = (float*)alloc((size_t)C_ * N1 * 4);
  _Float16* pxh     = (_Float16*)alloc((size_t)N1 * C_ * 2);
  _Float16* pyh     = (_Float16*)alloc((size_t)N1 * C_ * 2);

  dim3 w32(32);

  // ---- pre projection: nodes = pre_w @ inputs + pre_b ----
  build_h_inputs_kernel<<<(N1 * 32 + 127) / 128, 128, 0, stream>>>(inputs, h_in);
  pad_weight_kernel<<<(C_ * 32 + 127) / 128, 128, 0, stream>>>(pre_w, CIN_, 0, wtmp, C_, CIN_, 32);
  wmma_gemm_kernel<<<dim3(C_ / 16, N1 / 64), w32, 0, stream>>>(
      wtmp, h_in, C_, N1, 32, pre_b, 0, nodes, nullptr, 0, nullptr, nullptr, 0);

  // ---- residual dilated-conv stack ----
  for (int i = 0; i < 10; ++i) {
    const float* w1  = (const float*)d_in[4 + 7 * i + 0];
    const float* b1  = (const float*)d_in[4 + 7 * i + 1];
    const float* w2  = (const float*)d_in[4 + 7 * i + 2];
    const float* b2  = (const float*)d_in[4 + 7 * i + 3];
    const float* w3  = (const float*)d_in[4 + 7 * i + 4];
    const float* b3  = (const float*)d_in[4 + 7 * i + 5];
    const float* alp = (const float*)d_in[4 + 7 * i + 6];
    int dil = 1 << (i % 5);
    conv1d_k15_kernel<<<(32 * N1 + 127) / 128, 128, 0, stream>>>(
        nodes, w1, b1, h1, C_, 32, dil, 1, nullptr, nullptr);
    cvt_transpose_kernel<<<(32 * N1 + 255) / 256, 256, 0, stream>>>(h1, h1h, 32, N1);
    pad_weight_kernel<<<(32 * 32 + 127) / 128, 128, 0, stream>>>(w2, 32, 0, wtmp, 32, 32, 32);
    wmma_gemm_kernel<<<dim3(2, N1 / 64), w32, 0, stream>>>(
        wtmp, h1h, 32, N1, 32, b2, 1, h2, nullptr, 0, nullptr, nullptr, 0);
    conv1d_k15_kernel<<<(C_ * N1 + 127) / 128, 128, 0, stream>>>(
        h2, w3, b3, nodes, 32, C_, dil, 0, nodes, alp);
  }

  // ---- edge tensor construction ----
  build_h_data_kernel<<<(N1 * 96 + 127) / 128, 128, 0, stream>>>(nodes, inputs, h_data);
  pad_weight_kernel<<<(C_ * 96 + 127) / 128, 128, 0, stream>>>(edge_w, 268, 0, wtmp, C_, 84, 96);
  wmma_gemm_kernel<<<dim3(C_ / 16, N1 / 64), w32, 0, stream>>>(
      wtmp, h_data, C_, N1, 96, nullptr, 0, exf, nullptr, 0, nullptr, nullptr, 0);
  pad_weight_kernel<<<(C_ * 96 + 127) / 128, 128, 0, stream>>>(edge_w, 268, 84, wtmp, C_, 84, 96);
  wmma_gemm_kernel<<<dim3(C_ / 16, N1 / 64), w32, 0, stream>>>(
      wtmp, h_data, C_, N1, 96, nullptr, 0, eyf, nullptr, 0, nullptr, nullptr, 0);
  ppd_kernel<<<(C_ * L_ + 127) / 128, 128, 0, stream>>>(edge_w, ppd);
  build_edges_kernel<<<(C_ * N2 + 255) / 256, 256, 0, stream>>>(exf, eyf, ppd, edges);

  // ---- 12 attention/edge blocks ----
  for (int d = 0; d < D_; ++d) {
    const float* wq_d  = wq  + (size_t)d * 512 * C_;
    const float* wk_d  = wk  + (size_t)d * 512 * C_;
    const float* wv_d  = wv  + (size_t)d * 512 * C_;
    const float* wb_d  = wb  + (size_t)d * H_ * C_;
    const float* wo_d  = wo  + (size_t)d * C_ * 512;
    const float* bo_d  = bo  + (size_t)d * C_;
    const float* wi_d  = wi  + (size_t)d * C_ * C_;
    const float* wj_d  = wj  + (size_t)d * C_ * C_;
    const float* we1_d = we1 + (size_t)d * C_ * C_;
    const float* we2_d = we2 + (size_t)d * C_ * C_;
    const float* an_d  = a_n + d;
    const float* ae_d  = a_e + d;

    cvt_transpose_kernel<<<(C_ * N1 + 255) / 256, 256, 0, stream>>>(nodes, nodes_h, C_, N1);

    // q, k transposed f16 outs; v normal layout
    pad_weight_kernel<<<(512 * 64 + 127) / 128, 128, 0, stream>>>(wq_d, C_, 0, wtmp, 512, C_, 64);
    wmma_gemm_kernel<<<dim3(32, N1 / 64), w32, 0, stream>>>(
        wtmp, nodes_h, 512, N1, 64, nullptr, 0, nullptr, qT, 1, nullptr, nullptr, 0);
    pad_weight_kernel<<<(512 * 64 + 127) / 128, 128, 0, stream>>>(wk_d, C_, 0, wtmp, 512, C_, 64);
    wmma_gemm_kernel<<<dim3(32, N1 / 64), w32, 0, stream>>>(
        wtmp, nodes_h, 512, N1, 64, nullptr, 0, nullptr, kT, 1, nullptr, nullptr, 0);
    pad_weight_kernel<<<(512 * 64 + 127) / 128, 128, 0, stream>>>(wv_d, C_, 0, wtmp, 512, C_, 64);
    wmma_gemm_kernel<<<dim3(32, N1 / 64), w32, 0, stream>>>(
        wtmp, nodes_h, 512, N1, 64, nullptr, 0, nullptr, vh, 0, nullptr, nullptr, 0);

    // attention bias = wb @ edges
    cvt_transpose_kernel<<<(C_ * N2 + 255) / 256, 256, 0, stream>>>(edges, h_edges, C_, N2);
    pad_weight_kernel<<<(H_ * 64 + 127) / 128, 128, 0, stream>>>(wb_d, C_, 0, wtmp, H_, C_, 64);
    wmma_gemm_kernel<<<dim3(1, N2 / 64), w32, 0, stream>>>(
        wtmp, h_edges, H_, N2, 64, nullptr, 0, biasY, nullptr, 0, nullptr, nullptr, 0);

    // logits -> softmax -> context
    attn_logits_kernel<<<dim3(B_ * H_, L_ / 16, L_ / 64), w32, 0, stream>>>(
        qT, kT, biasY, mask, logits);
    softmax_kernel<<<B_ * H_ * L_, w32, 0, stream>>>(logits, attn_h);
    attn_ctx_kernel<<<dim3(B_ * H_, VS_ / 16, L_ / 64), w32, 0, stream>>>(attn_h, vh, ctxT);

    // nodes += a_n * (wo @ ctx + bo)
    pad_weight_kernel<<<(C_ * 512 + 127) / 128, 128, 0, stream>>>(wo_d, 512, 0, wtmp, C_, 512, 512);
    wmma_gemm_kernel<<<dim3(C_ / 16, N1 / 64), w32, 0, stream>>>(
        wtmp, ctxT, C_, N1, 512, bo_d, 0, nodes, nullptr, 0, nodes, an_d, 0);

    // ni = wi @ nodes, nj = wj @ nodes
    cvt_transpose_kernel<<<(C_ * N1 + 255) / 256, 256, 0, stream>>>(nodes, nodes_h, C_, N1);
    pad_weight_kernel<<<(C_ * C_ + 127) / 128, 128, 0, stream>>>(wi_d, C_, 0, wtmp, C_, C_, 64);
    wmma_gemm_kernel<<<dim3(C_ / 16, N1 / 64), w32, 0, stream>>>(
        wtmp, nodes_h, C_, N1, 64, nullptr, 0, nif, nullptr, 0, nullptr, nullptr, 0);
    pad_weight_kernel<<<(C_ * C_ + 127) / 128, 128, 0, stream>>>(wj_d, C_, 0, wtmp, C_, C_, 64);
    wmma_gemm_kernel<<<dim3(C_ / 16, N1 / 64), w32, 0, stream>>>(
        wtmp, nodes_h, C_, N1, 64, nullptr, 0, njf, nullptr, 0, nullptr, nullptr, 0);

    // edge MLP: edges += a_e * we2 @ relu(we1 @ (edges + ni + nj))
    build_t_kernel<<<(C_ * N2 + 255) / 256, 256, 0, stream>>>(edges, nif, njf, t_h);
    pad_weight_kernel<<<(C_ * C_ + 127) / 128, 128, 0, stream>>>(we1_d, C_, 0, wtmp, C_, C_, 64);
    wmma_gemm_kernel<<<dim3(C_ / 16, N2 / 64), w32, 0, stream>>>(
        wtmp, t_h, C_, N2, 64, nullptr, 1, nullptr, hbufT, 1, nullptr, nullptr, 0);
    pad_weight_kernel<<<(C_ * C_ + 127) / 128, 128, 0, stream>>>(we2_d, C_, 0, wtmp, C_, C_, 64);
    wmma_gemm_kernel<<<dim3(C_ / 16, N2 / 64), w32, 0, stream>>>(
        wtmp, hbufT, C_, N2, 64, nullptr, 0, edges, nullptr, 0, edges, ae_d, 0);
  }

  // ---- prediction heads ----
  float* out = (float*)d_out;
  cvt_transpose_kernel<<<(C_ * N1 + 255) / 256, 256, 0, stream>>>(nodes, nodes_h, C_, N1);

  const int seqO[3] = {36, 36, 20};
  size_t soff = 0;
  for (int j = 0; j < 3; ++j) {
    const float* w  = (const float*)d_in[87 + 2 * j];
    const float* bb = (const float*)d_in[88 + 2 * j];
    pad_weight_kernel<<<(seqO[j] * 64 + 127) / 128, 128, 0, stream>>>(w, C_, 0, wtmp, seqO[j], C_, 64);
    wmma_gemm_kernel<<<dim3((seqO[j] + 15) / 16, N1 / 64), w32, 0, stream>>>(
        wtmp, nodes_h, seqO[j], N1, 64, bb, 0, out + soff, nullptr, 0, nullptr, nullptr, 1);
    soff += (size_t)B_ * seqO[j] * L_;
  }

  cvt_transpose_kernel<<<(C_ * N2 + 255) / 256, 256, 0, stream>>>(edges, h_edges, C_, N2);
  const int pairO[4] = {42, 36, 36, 36};
  size_t poff = soff;
  for (int j = 0; j < 4; ++j) {
    const float* w  = (const float*)d_in[93 + 2 * j];
    const float* bb = (const float*)d_in[94 + 2 * j];
    pad_weight_kernel<<<(pairO[j] * 64 + 127) / 128, 128, 0, stream>>>(w, C_, 0, wtmp, pairO[j], C_, 64);
    wmma_gemm_kernel<<<dim3((pairO[j] + 15) / 16, N2 / 64), w32, 0, stream>>>(
        wtmp, h_edges, pairO[j], N2, 64, bb, 0, out + poff, nullptr, 0, nullptr, nullptr, 2);
    poff += (size_t)B_ * pairO[j] * NLL;
  }

  mean_last_kernel<<<(C_ * N1 + 127) / 128, 128, 0, stream>>>(edges, pxf);
  mean_mid_kernel<<<(C_ * N1 + 127) / 128, 128, 0, stream>>>(edges, pyf);
  cvt_transpose_kernel<<<(C_ * N1 + 255) / 256, 256, 0, stream>>>(pxf, pxh, C_, N1);
  cvt_transpose_kernel<<<(C_ * N1 + 255) / 256, 256, 0, stream>>>(pyf, pyh, C_, N1);

  const int projO[3] = {36, 36, 20};
  size_t xoff = poff;
  for (int j = 0; j < 3; ++j) {
    const float* w  = (const float*)d_in[101 + 2 * j];
    const float* bb = (const float*)d_in[102 + 2 * j];
    pad_weight_kernel<<<(projO[j] * 64 + 127) / 128, 128, 0, stream>>>(w, C_, 0, wtmp, projO[j], C_, 64);
    wmma_gemm_kernel<<<dim3((projO[j] + 15) / 16, N1 / 64), w32, 0, stream>>>(
        wtmp, pxh, projO[j], N1, 64, bb, 0, out + xoff, nullptr, 0, nullptr, nullptr, 1);
    xoff += (size_t)B_ * projO[j] * L_;
  }
  size_t yoff = xoff;
  for (int j = 0; j < 3; ++j) {
    const float* w  = (const float*)d_in[101 + 2 * j];
    const float* bb = (const float*)d_in[102 + 2 * j];
    pad_weight_kernel<<<(projO[j] * 64 + 127) / 128, 128, 0, stream>>>(w, C_, 0, wtmp, projO[j], C_, 64);
    wmma_gemm_kernel<<<dim3((projO[j] + 15) / 16, N1 / 64), w32, 0, stream>>>(
        wtmp, pyh, projO[j], N1, 64, bb, 0, out + yoff, nullptr, 0, nullptr, nullptr, 1);
    yoff += (size_t)B_ * projO[j] * L_;
  }
}
